// GCN_Encoder_55035710931806
// MI455X (gfx1250) — compile-verified, matrix operand order checked by reading, not run on hardware
//
#include <hip/hip_runtime.h>
#include <hip/hip_bf16.h>

typedef __attribute__((ext_vector_type(16))) __bf16 v16bf;
typedef __attribute__((ext_vector_type(8)))  float  v8f;

#define DIM 128
#define LEAKY 0.01f
#define BN_EPS 1e-5f

// ---------------------------------------------------------------------------
// GEMM: out[N,128] = H(bf16)[N,128] @ W + b, W given pre-transposed bf16 WT[n][k]
// block = 128 threads (4 waves) -> 16 rows; each wave owns 2 col-tiles.
// Single K-loop, A fragment shared by both col-tiles (2 WMMAs per K step).
// ---------------------------------------------------------------------------
__global__ void __launch_bounds__(128)
gcn_gemm_wmma(const __bf16* __restrict__ H, const __bf16* __restrict__ WT,
              const float* __restrict__ bias, float* __restrict__ out, int nrows)
{
    const int wave = threadIdx.x >> 5;
    const int lane = threadIdx.x & 31;
    const int half = lane >> 4;
    const int l15  = lane & 15;
    const int rowBase = blockIdx.x * 16;

    int arow = rowBase + l15;
    if (arow >= nrows) arow = nrows - 1;                 // clamped load (stores guarded)
    const unsigned int* hrow = (const unsigned int*)(H + (size_t)arow * DIM); // bf16 pairs

    const int nBase0 = wave * 32;
    const int nBase1 = wave * 32 + 16;
    const unsigned int* wt0 = (const unsigned int*)(WT + (size_t)(nBase0 + l15) * DIM);
    const unsigned int* wt1 = (const unsigned int*)(WT + (size_t)(nBase1 + l15) * DIM);

    v8f c0, c1;
    const float bv0 = bias[nBase0 + l15];
    const float bv1 = bias[nBase1 + l15];
    #pragma unroll
    for (int r = 0; r < 8; ++r) { c0[r] = bv0; c1[r] = bv1; }  // C init = bias

    #pragma unroll
    for (int kk = 0; kk < DIM; kk += 32) {
        union { v16bf v; unsigned int u[8]; } a, b0, b1;
        const int kp = kk >> 1;                          // bf16-pair index base
        #pragma unroll
        for (int r = 0; r < 4; ++r) {
            // 16-bit 16x32 layout: VGPR r: K=2r(+8*half); VGPR r+4: K=16+2r(+8*half)
            a.u[r]      = hrow[kp + r + 4 * half];
            a.u[r + 4]  = hrow[kp + 8 + r + 4 * half];
            b0.u[r]     = wt0 [kp + r + 4 * half];
            b0.u[r + 4] = wt0 [kp + 8 + r + 4 * half];
            b1.u[r]     = wt1 [kp + r + 4 * half];
            b1.u[r + 4] = wt1 [kp + 8 + r + 4 * half];
        }
        c0 = __builtin_amdgcn_wmma_f32_16x16x32_bf16(
                 false, a.v, false, b0.v, (short)0, c0, false, false);
        c1 = __builtin_amdgcn_wmma_f32_16x16x32_bf16(
                 false, a.v, false, b1.v, (short)0, c1, false, false);
    }

    // C/D layout: VGPR r holds (M = r + 8*half, N = lane&15)
    if (rowBase + 16 <= nrows) {                         // uniform: fast full-tile path
        #pragma unroll
        for (int r = 0; r < 8; ++r) {
            const size_t m = (size_t)(rowBase + r + 8 * half);
            out[m * DIM + nBase0 + l15] = c0[r];
            out[m * DIM + nBase1 + l15] = c1[r];
        }
    } else {
        #pragma unroll
        for (int r = 0; r < 8; ++r) {
            const int m = rowBase + r + 8 * half;
            if (m < nrows) {
                out[(size_t)m * DIM + nBase0 + l15] = c0[r];
                out[(size_t)m * DIM + nBase1 + l15] = c1[r];
            }
        }
    }
}

// ---------------------------------------------------------------------------
// Prep / elementwise kernels
// ---------------------------------------------------------------------------
__global__ void k_f32_to_bf16(const float* __restrict__ in, __bf16* __restrict__ out,
                              long long n)
{
    for (long long i = (long long)blockIdx.x * blockDim.x + threadIdx.x; i < n;
         i += (long long)gridDim.x * blockDim.x)
        out[i] = (__bf16)in[i];
}

__global__ void k_transpose_w(const float* __restrict__ Ws, __bf16* __restrict__ WT,
                              int total)  // total = L*128*128
{
    int i = blockIdx.x * blockDim.x + threadIdx.x;
    if (i >= total) return;
    int l  = i >> 14;            // /16384
    int r  = i & 16383;
    int k  = r >> 7;
    int nn = r & 127;
    WT[(size_t)l * 16384 + (size_t)nn * DIM + k] = (__bf16)Ws[i]; // i = l*16384 + k*128 + nn
}

__global__ void k_fill_f32(float* __restrict__ p, float v, long long n)
{
    for (long long i = (long long)blockIdx.x * blockDim.x + threadIdx.x; i < n;
         i += (long long)gridDim.x * blockDim.x)
        p[i] = v;
}

__global__ void k_deg_edges(const int* __restrict__ ei, int E, float* __restrict__ deg)
{
    int e = blockIdx.x * blockDim.x + threadIdx.x;
    if (e < E) atomicAdd(&deg[ei[E + e]], 1.0f);
}

__global__ void k_rsqrt_deg(float* __restrict__ p, int n)
{
    int i = blockIdx.x * blockDim.x + threadIdx.x;
    if (i < n) p[i] = rsqrtf(fmaxf(p[i], 1.0f));
}

// acc[i,:] = t[i,:] * dinv[i]^2  (self-loop term; also zero-initializes acc)
__global__ void k_selfloop_init(const float* __restrict__ t, const float* __restrict__ dinv,
                                float* __restrict__ acc, int n)
{
    long long idx = (long long)blockIdx.x * blockDim.x + threadIdx.x;
    long long total = (long long)n * 32;
    if (idx >= total) return;
    int node = (int)(idx >> 5);
    int q    = (int)(idx & 31);
    float d = dinv[node];
    float s = d * d;
    float4 v = ((const float4*)(t + (size_t)node * DIM))[q];
    v.x *= s; v.y *= s; v.z *= s; v.w *= s;
    ((float4*)(acc + (size_t)node * DIM))[q] = v;
}

// 32 lanes per edge, float4 gather + 4 atomic f32 adds each
__global__ void k_edge_scatter(const int* __restrict__ ei, int E,
                               const float* __restrict__ t, const float* __restrict__ dinv,
                               float* __restrict__ acc)
{
    long long gid = (long long)blockIdx.x * blockDim.x + threadIdx.x;
    long long total = (long long)E * 32;
    if (gid >= total) return;
    int e = (int)(gid >> 5);
    int q = (int)(gid & 31);
    int s = ei[e];
    int d = ei[E + e];
    float nrm = dinv[s] * dinv[d];
    float4 v = ((const float4*)(t + (size_t)s * DIM))[q];
    float* o = acc + (size_t)d * DIM + q * 4;
    atomicAdd(o + 0, v.x * nrm);
    atomicAdd(o + 1, v.y * nrm);
    atomicAdd(o + 2, v.z * nrm);
    atomicAdd(o + 3, v.w * nrm);
}

// LeakyReLU in place + per-column sum / sumsq -> stats[0..127]=sum, [128..255]=sumsq
__global__ void __launch_bounds__(256)
k_stats_leaky(float* __restrict__ h, int n, float* __restrict__ stats)
{
    __shared__ float lsum[DIM];
    __shared__ float lsq[DIM];
    for (int i = threadIdx.x; i < DIM; i += blockDim.x) { lsum[i] = 0.f; lsq[i] = 0.f; }
    __syncthreads();

    const int c = (threadIdx.x & 31) * 4;   // stride keeps (idx&31) invariant
    float s0=0,s1=0,s2=0,s3=0,q0=0,q1=0,q2=0,q3=0;
    long long total = (long long)n * 32;
    for (long long idx = (long long)blockIdx.x * blockDim.x + threadIdx.x; idx < total;
         idx += (long long)gridDim.x * blockDim.x) {
        float4* p = ((float4*)h) + idx;
        float4 v = *p;
        v.x = v.x > 0.f ? v.x : v.x * LEAKY;
        v.y = v.y > 0.f ? v.y : v.y * LEAKY;
        v.z = v.z > 0.f ? v.z : v.z * LEAKY;
        v.w = v.w > 0.f ? v.w : v.w * LEAKY;
        *p = v;
        s0 += v.x; q0 += v.x * v.x;
        s1 += v.y; q1 += v.y * v.y;
        s2 += v.z; q2 += v.z * v.z;
        s3 += v.w; q3 += v.w * v.w;
    }
    atomicAdd(&lsum[c + 0], s0); atomicAdd(&lsq[c + 0], q0);
    atomicAdd(&lsum[c + 1], s1); atomicAdd(&lsq[c + 1], q1);
    atomicAdd(&lsum[c + 2], s2); atomicAdd(&lsq[c + 2], q2);
    atomicAdd(&lsum[c + 3], s3); atomicAdd(&lsq[c + 3], q3);
    __syncthreads();
    if (threadIdx.x < DIM)       atomicAdd(&stats[threadIdx.x], lsum[threadIdx.x]);
    else if (threadIdx.x < 256)  atomicAdd(&stats[threadIdx.x], lsq[threadIdx.x - DIM]);
}

// BN apply in place (f32, used for pooling) + write bf16 copy for next layer's GEMM
__global__ void k_bn_apply(float* __restrict__ h, __bf16* __restrict__ hbf, int n,
                           const float* __restrict__ stats,
                           const float* __restrict__ gamma, const float* __restrict__ beta,
                           float inv_n)
{
    long long idx = (long long)blockIdx.x * blockDim.x + threadIdx.x;
    long long total = (long long)n * 32;
    if (idx >= total) return;
    const int c = (int)(idx & 31) * 4;

    float mu0 = stats[c + 0] * inv_n, mu1 = stats[c + 1] * inv_n;
    float mu2 = stats[c + 2] * inv_n, mu3 = stats[c + 3] * inv_n;
    float r0 = rsqrtf(stats[DIM + c + 0] * inv_n - mu0 * mu0 + BN_EPS);
    float r1 = rsqrtf(stats[DIM + c + 1] * inv_n - mu1 * mu1 + BN_EPS);
    float r2 = rsqrtf(stats[DIM + c + 2] * inv_n - mu2 * mu2 + BN_EPS);
    float r3 = rsqrtf(stats[DIM + c + 3] * inv_n - mu3 * mu3 + BN_EPS);

    float4 v = ((float4*)h)[idx];
    v.x = (v.x - mu0) * r0 * gamma[c + 0] + beta[c + 0];
    v.y = (v.y - mu1) * r1 * gamma[c + 1] + beta[c + 1];
    v.z = (v.z - mu2) * r2 * gamma[c + 2] + beta[c + 2];
    v.w = (v.w - mu3) * r3 * gamma[c + 3] + beta[c + 3];
    ((float4*)h)[idx] = v;

    __bf16* bp = hbf + (size_t)idx * 4;
    bp[0] = (__bf16)v.x; bp[1] = (__bf16)v.y; bp[2] = (__bf16)v.z; bp[3] = (__bf16)v.w;
}

// ---------------------------------------------------------------------------
// Pooling
// ---------------------------------------------------------------------------
__global__ void k_pool_acc(const float* __restrict__ h, const int* __restrict__ batch,
                           float* __restrict__ pool, float* __restrict__ cnt, int n)
{
    long long idx = (long long)blockIdx.x * blockDim.x + threadIdx.x;
    long long total = (long long)n * 32;
    if (idx >= total) return;
    int node = (int)(idx >> 5);
    int q    = (int)(idx & 31);
    int g = batch[node];
    float4 v = ((const float4*)(h + (size_t)node * DIM))[q];
    float* o = pool + (size_t)g * DIM + q * 4;
    atomicAdd(o + 0, v.x); atomicAdd(o + 1, v.y);
    atomicAdd(o + 2, v.z); atomicAdd(o + 3, v.w);
    if (q == 0) atomicAdd(&cnt[g], 1.0f);
}

__global__ void k_pool_final(const float* __restrict__ pool, const float* __restrict__ cnt,
                             float* __restrict__ out, int total)
{
    int i = blockIdx.x * blockDim.x + threadIdx.x;
    if (i >= total) return;
    out[i] = pool[i] / fmaxf(cnt[i >> 7], 1.0f);
}

// ---------------------------------------------------------------------------
// Host launcher
// ---------------------------------------------------------------------------
static inline size_t align256(size_t x) { return (x + 255) & ~(size_t)255; }

extern "C" void kernel_launch(void* const* d_in, const int* in_sizes, int n_in,
                              void* d_out, int out_size, void* d_ws, size_t ws_size,
                              hipStream_t stream)
{
    const float* x      = (const float*)d_in[0];
    const int*   ei     = (const int*)  d_in[1];
    const int*   batch  = (const int*)  d_in[2];
    const float* Ws     = (const float*)d_in[3];
    const float* bs     = (const float*)d_in[4];
    const float* gammas = (const float*)d_in[5];
    const float* betas  = (const float*)d_in[6];

    const int N = in_sizes[0] / DIM;
    const int E = in_sizes[1] / 2;
    const int L = in_sizes[3] / (DIM * DIM);
    const int G = out_size / DIM;

    // workspace partition
    char* w = (char*)d_ws;
    float*  t     = (float*)w;  w += align256((size_t)N * DIM * 4);
    float*  hacc  = (float*)w;  w += align256((size_t)N * DIM * 4);
    __bf16* hbf   = (__bf16*)w; w += align256((size_t)N * DIM * 2);
    __bf16* wt    = (__bf16*)w; w += align256((size_t)L * DIM * DIM * 2);
    float*  dinv  = (float*)w;  w += align256((size_t)N * 4);
    float*  stats = (float*)w;  w += align256(256 * 4);
    float*  pool  = (float*)w;  w += align256((size_t)G * DIM * 4);
    float*  cnt   = (float*)w;  w += align256((size_t)G * 4);

    const long long ND   = (long long)N * DIM;
    const long long N32  = (long long)N * 32;
    const long long E32  = (long long)E * 32;
    const int B = 256;

    // --- prep ---
    k_f32_to_bf16 <<<2048, B, 0, stream>>>(x, hbf, ND);
    k_transpose_w <<<(L * DIM * DIM + B - 1) / B, B, 0, stream>>>(Ws, wt, L * DIM * DIM);
    k_fill_f32    <<<(int)((N + B - 1) / B), B, 0, stream>>>(dinv, 1.0f, N);   // self-loop deg
    k_deg_edges   <<<(E + B - 1) / B, B, 0, stream>>>(ei, E, dinv);
    k_rsqrt_deg   <<<(N + B - 1) / B, B, 0, stream>>>(dinv, N);

    // --- 7 GCN layers ---
    for (int l = 0; l < L; ++l) {
        gcn_gemm_wmma  <<<(N + 15) / 16, 128, 0, stream>>>(hbf, wt + (size_t)l * DIM * DIM,
                                                           bs + l * DIM, t, N);
        k_selfloop_init<<<(int)((N32 + B - 1) / B), B, 0, stream>>>(t, dinv, hacc, N);
        k_edge_scatter <<<(int)((E32 + B - 1) / B), B, 0, stream>>>(ei, E, t, dinv, hacc);
        k_fill_f32     <<<1, B, 0, stream>>>(stats, 0.0f, 256);
        k_stats_leaky  <<<1024, B, 0, stream>>>(hacc, N, stats);
        k_bn_apply     <<<(int)((N32 + B - 1) / B), B, 0, stream>>>(hacc, hbf, N, stats,
                                                                    gammas + l * DIM,
                                                                    betas + l * DIM,
                                                                    1.0f / (float)N);
    }

    // --- global mean pool ---
    k_fill_f32  <<<(G * DIM + G + B - 1) / B, B, 0, stream>>>(pool, 0.0f, (long long)G * DIM + G);
    k_pool_acc  <<<(int)((N32 + B - 1) / B), B, 0, stream>>>(hacc, batch, pool, cnt, N);
    k_pool_final<<<(G * DIM + B - 1) / B, B, 0, stream>>>(pool, cnt, (float*)d_out, G * DIM);
}
